// Attention_9766755631105
// MI455X (gfx1250) — matrix operand order repaired and checked by reading, not compiled, not run on hardware
//
#include <hip/hip_runtime.h>
#include <hip/hip_bf16.h>
#include <stdint.h>

// Problem constants (match reference)
#define B_  4
#define L_  2048
#define D_  1024
#define H_  16
#define HD_ 64
#define M_  (B_ * L_)    // 8192 rows
#define N3_ (3 * D_)     // 3072 qkv cols

typedef __bf16 v16bf __attribute__((ext_vector_type(16)));
typedef __bf16 v2bf  __attribute__((ext_vector_type(2)));
typedef float  v8f   __attribute__((ext_vector_type(8)));
typedef float  f4    __attribute__((ext_vector_type(4)));
typedef int    v4i   __attribute__((ext_vector_type(4)));

#if defined(__has_builtin)
#if __has_builtin(__builtin_amdgcn_global_load_async_to_lds_b128)
#define HAVE_ASYNC_LDS 1
#endif
#endif
#ifndef HAVE_ASYNC_LDS
#define HAVE_ASYNC_LDS 0
#endif

union Frag { uint32_t u[8]; v16bf v; };

// One v_cvt_pk_bf16_f32 per bf16 pair: both halves of the dst VGPR in one op.
// (clang scalarizes the builtin + union path into cvt+v_mov_b16 chains.)
__device__ __forceinline__ uint32_t pack2(float lo, float hi) {
  uint32_t r;
  asm("v_cvt_pk_bf16_f32 %0, %1, %2" : "=v"(r) : "v"(lo), "v"(hi));
  return r;
}
__device__ __forceinline__ uint16_t bf16_bits(float f) {
  union { __bf16 h; uint16_t u; } r;
  r.h = (__bf16)f;
  return r.u;
}

// ---------------------------------------------------------------------------
// 16x64 output tile GEMM core: C[m, n..n+63] = sum_k X[m,k] * W[n,k]
// X row-major [M,K] f32, W row-major [N,K] f32; bf16 WMMA, f32 accumulate.
// A frag (16x32): lane m=lane%16; half0 K=0..15, half1 K=16..31
// B frag (32x16): lane n=lane%16; half0 K=0..15, half1 K=16..31 (pairs/VGPR)
// One A fragment is reused across 4 WMMAs (4 N-subtiles).
// ---------------------------------------------------------------------------
__device__ __forceinline__ void gemm16x64(const float* __restrict__ xrow,
                                          const float* __restrict__ wrow,
                                          int K, int half, v8f c[4]) {
  for (int k0 = 0; k0 < K; k0 += 32) {
    Frag a;
    const float* ap = xrow + k0 + half * 16;
    f4 a0 = *(const f4*)(ap);
    f4 a1 = *(const f4*)(ap + 4);
    f4 a2 = *(const f4*)(ap + 8);
    f4 a3 = *(const f4*)(ap + 12);
    a.u[0] = pack2(a0[0], a0[1]); a.u[1] = pack2(a0[2], a0[3]);
    a.u[2] = pack2(a1[0], a1[1]); a.u[3] = pack2(a1[2], a1[3]);
    a.u[4] = pack2(a2[0], a2[1]); a.u[5] = pack2(a2[2], a2[3]);
    a.u[6] = pack2(a3[0], a3[1]); a.u[7] = pack2(a3[2], a3[3]);
    if (k0 + 32 < K) {                       // prefetch next k-step (global_prefetch_b8)
      __builtin_prefetch(xrow + k0 + 32, 0, 3);
      __builtin_prefetch(wrow + k0 + 32, 0, 3);
    }
#pragma unroll
    for (int nt = 0; nt < 4; ++nt) {
      const float* bp = wrow + (size_t)nt * 16 * K + k0 + half * 16;
      Frag b;
      f4 b0 = *(const f4*)(bp);
      f4 b1 = *(const f4*)(bp + 4);
      f4 b2 = *(const f4*)(bp + 8);
      f4 b3 = *(const f4*)(bp + 12);
      b.u[0] = pack2(b0[0], b0[1]); b.u[1] = pack2(b0[2], b0[3]);
      b.u[2] = pack2(b1[0], b1[1]); b.u[3] = pack2(b1[2], b1[3]);
      b.u[4] = pack2(b2[0], b2[1]); b.u[5] = pack2(b2[2], b2[3]);
      b.u[6] = pack2(b3[0], b3[1]); b.u[7] = pack2(b3[2], b3[3]);
      c[nt] = __builtin_amdgcn_wmma_f32_16x16x32_bf16(false, a.v, false, b.v,
                                                      (short)0, c[nt], false, false);
    }
  }
}

// ---------------------------------------------------------------------------
// Kernel 1: qkv = x @ Wqkv^T + bqkv, written bf16 into head-split layout:
//   qkvb[which][b][h][l][d],  which in {q,k,v}
// ---------------------------------------------------------------------------
__global__ __launch_bounds__(256)
void qkv_gemm_kernel(const float* __restrict__ X, const float* __restrict__ W,
                     const float* __restrict__ bias, uint16_t* __restrict__ qkvb) {
  int wave = (int)((blockIdx.x * blockDim.x + threadIdx.x) >> 5);
  int lane = threadIdx.x & 31;
  const int ntTiles = N3_ / 64;             // 48
  int tm = (wave / ntTiles) * 16;
  int tn = (wave % ntTiles) * 64;
  int col = lane & 15, half = lane >> 4;

  v8f c[4] = {};
  gemm16x64(X + (size_t)(tm + col) * D_, W + (size_t)(tn + col) * D_, D_, half, c);

#pragma unroll
  for (int nt = 0; nt < 4; ++nt) {
    int e      = tn + nt * 16 + col;        // column in [0, 3D)
    int which  = e >> 10;                   // 0=q 1=k 2=v
    int within = e & (D_ - 1);
    int h = within >> 6, d = within & (HD_ - 1);
    float bv = bias[e];
    size_t base = (size_t)which * ((size_t)B_ * H_ * L_ * HD_);
#pragma unroll
    for (int r = 0; r < 8; ++r) {
      int row = tm + half * 8 + r;          // global token row (b*L + l)
      int bb = row >> 11, l = row & (L_ - 1);
      qkvb[base + ((((size_t)bb * H_ + h) * L_ + l) * HD_ + d)] =
          bf16_bits(c[nt][r] + bv);
    }
  }
}

// ---------------------------------------------------------------------------
// Kernel 2: causal flash attention. One wave per (b,h, 16-query tile).
// K/V chunks staged into LDS with async-to-LDS copies (ASYNCcnt) when available.
// ---------------------------------------------------------------------------
__global__ __launch_bounds__(128)
void attn_kernel(const uint16_t* __restrict__ Qb, const uint16_t* __restrict__ Kb,
                 const uint16_t* __restrict__ Vb, float* __restrict__ AO) {
  __shared__ uint16_t plds[4][16 * 32];     // per-wave P staging (1KB each)
#if HAVE_ASYNC_LDS
  __shared__ uint16_t kvlds[4][2][32 * HD_]; // per-wave K/V chunk (4KB each)
#endif
  int wid  = threadIdx.x >> 5;
  int lane = threadIdx.x & 31;
  int wave = blockIdx.x * 4 + wid;
  const int nQT = L_ / 16;                  // 128
  int qt = wave % nQT;
  int bh = wave / nQT;                      // b*H + h
  int qBase = qt * 16;
  int col = lane & 15, half = lane >> 4;

  const uint16_t* Qp = Qb + ((size_t)bh * L_ + qBase) * HD_;
  const uint16_t* Kp = Kb + (size_t)bh * L_ * HD_;
  const uint16_t* Vp = Vb + (size_t)bh * L_ * HD_;

  // Q as two 16x32 A-fragments (d 0..31 and 32..63), resident for whole loop.
  Frag qa[2];
  const uint16_t* qrow = Qp + col * HD_;
#pragma unroll
  for (int t = 0; t < 2; ++t) {
    int db = t * 32 + half * 16;
#pragma unroll
    for (int i = 0; i < 4; ++i)
      qa[t].u[i]     = *(const uint32_t*)(qrow + db + 2 * i);
#pragma unroll
    for (int i = 0; i < 4; ++i)
      qa[t].u[4 + i] = *(const uint32_t*)(qrow + db + 8 + 2 * i);
  }

  float m[8], lsum[8], alpha[8];
  v8f o[4] = {};                            // 16x64 output accumulator (4 C tiles)
#pragma unroll
  for (int r = 0; r < 8; ++r) { m[r] = -1e30f; lsum[r] = 0.f; }

  const float scale = 0.125f;               // 1/sqrt(64)
  uint16_t* lds = plds[wid];

  for (int kb = 0; kb < qBase + 16; kb += 32) {   // causal: keys <= qBase+15
#if HAVE_ASYNC_LDS
    // Previous chunk's LDS reads must retire before we overwrite the buffer.
    asm volatile("s_wait_dscnt 0" ::: "memory");
    {
      const char* gK = (const char*)(Kp + (size_t)kb * HD_); // 4KB contiguous
      const char* gV = (const char*)(Vp + (size_t)kb * HD_);
      char* lK = (char*)&kvlds[wid][0][0];
      char* lV = (char*)&kvlds[wid][1][0];
#pragma unroll
      for (int i = 0; i < 8; ++i) {
        int off = (lane + 32 * i) * 16;
        __builtin_amdgcn_global_load_async_to_lds_b128(
            (__attribute__((address_space(1))) v4i*)(gK + off),
            (__attribute__((address_space(3))) v4i*)(lK + off), 0, 0);
        __builtin_amdgcn_global_load_async_to_lds_b128(
            (__attribute__((address_space(1))) v4i*)(gV + off),
            (__attribute__((address_space(3))) v4i*)(lV + off), 0, 0);
      }
    }
#if defined(__has_builtin) && __has_builtin(__builtin_amdgcn_s_wait_asynccnt)
    __builtin_amdgcn_s_wait_asynccnt(0);
#else
    asm volatile("s_wait_asynccnt 0" ::: "memory");
#endif
    __builtin_amdgcn_wave_barrier();
    const uint16_t* Kchunk = &kvlds[wid][0][0];   // local keys 0..31
    const uint16_t* Vchunk = &kvlds[wid][1][0];
#else
    const uint16_t* Kchunk = Kp + (size_t)kb * HD_;
    const uint16_t* Vchunk = Vp + (size_t)kb * HD_;
#endif

    // ---- S = Q K^T for two 16-key tiles (K^T B-frags: contiguous d per key row)
    v8f s[2];
#pragma unroll
    for (int kt = 0; kt < 2; ++kt) {
      const uint16_t* krow = Kchunk + (size_t)(kt * 16 + col) * HD_;
      Frag b0, b1;
#pragma unroll
      for (int i = 0; i < 8; ++i)
        b0.u[i] = *(const uint32_t*)(krow + half * 16 + 2 * i);
#pragma unroll
      for (int i = 0; i < 8; ++i)
        b1.u[i] = *(const uint32_t*)(krow + 32 + half * 16 + 2 * i);
      v8f acc = {};
      acc = __builtin_amdgcn_wmma_f32_16x16x32_bf16(false, qa[0].v, false, b0.v,
                                                    (short)0, acc, false, false);
      acc = __builtin_amdgcn_wmma_f32_16x16x32_bf16(false, qa[1].v, false, b1.v,
                                                    (short)0, acc, false, false);
      s[kt] = acc;
    }

    // ---- online softmax (row r of tile lives at VGPR r across 16 lanes/half)
#pragma unroll
    for (int r = 0; r < 8; ++r) {
      int row = qBase + half * 8 + r;
      int j0 = kb + col, j1 = kb + 16 + col;
      float x0 = (j0 <= row) ? s[0][r] * scale : -1e30f;
      float x1 = (j1 <= row) ? s[1][r] * scale : -1e30f;
      float mx = fmaxf(x0, x1);
#pragma unroll
      for (int off = 1; off < 16; off <<= 1)
        mx = fmaxf(mx, __shfl_xor(mx, off, 32));
      float mn = fmaxf(m[r], mx);
      float al = __expf(m[r] - mn);
      float p0 = __expf(x0 - mn);
      float p1 = __expf(x1 - mn);
      float ps = p0 + p1;
#pragma unroll
      for (int off = 1; off < 16; off <<= 1)
        ps += __shfl_xor(ps, off, 32);
      lsum[r] = lsum[r] * al + ps;
      m[r] = mn; alpha[r] = al;
      int rl = half * 8 + r;                // write P row-major [q_local][key_local]
      lds[rl * 32 + col]      = bf16_bits(p0);
      lds[rl * 32 + 16 + col] = bf16_bits(p1);
    }
    __builtin_amdgcn_wave_barrier();
    asm volatile("s_wait_dscnt 0" ::: "memory");   // cross-lane LDS visibility
    __builtin_amdgcn_wave_barrier();

    // ---- P (C-layout) -> A-fragment via LDS transpose
    Frag pa;
    const uint16_t* prow = lds + col * 32;
#pragma unroll
    for (int i = 0; i < 4; ++i)
      pa.u[i]     = *(const uint32_t*)(prow + half * 16 + 2 * i);
#pragma unroll
    for (int i = 0; i < 4; ++i)
      pa.u[4 + i] = *(const uint32_t*)(prow + half * 16 + 8 + 2 * i);
    __builtin_amdgcn_wave_barrier();

    // ---- O = O*alpha + P @ V   (V rows are keys; B-frag needs key-pairs per VGPR)
#pragma unroll
    for (int t = 0; t < 4; ++t) {
#pragma unroll
      for (int r = 0; r < 8; ++r) o[t][r] *= alpha[r];
      Frag vb;
#pragma unroll
      for (int i = 0; i < 8; ++i) {
        int k = half * 16 + 2 * i;
        uint32_t lo = Vchunk[(size_t)k       * HD_ + t * 16 + col];
        uint32_t hi = Vchunk[(size_t)(k + 1) * HD_ + t * 16 + col];
        vb.u[i] = lo | (hi << 16);
      }
      o[t] = __builtin_amdgcn_wmma_f32_16x16x32_bf16(false, pa.v, false, vb.v,
                                                     (short)0, o[t], false, false);
    }
  }

  // ---- epilogue: normalize and store f32 in [b, l, h*64+d] layout
  int b = bh / H_, h = bh % H_;
#pragma unroll
  for (int r = 0; r < 8; ++r) {
    int row = qBase + half * 8 + r;
    float inv = 1.0f / lsum[r];
#pragma unroll
    for (int t = 0; t < 4; ++t)
      AO[((size_t)(b * L_ + row)) * D_ + h * HD_ + t * 16 + col] = o[t][r] * inv;
  }
}

// ---------------------------------------------------------------------------
// Kernel 3: out = attnOut @ Wo^T + bo  (f32 output)
// ---------------------------------------------------------------------------
__global__ __launch_bounds__(256)
void out_gemm_kernel(const float* __restrict__ X, const float* __restrict__ W,
                     const float* __restrict__ bias, float* __restrict__ out) {
  int wave = (int)((blockIdx.x * blockDim.x + threadIdx.x) >> 5);
  int lane = threadIdx.x & 31;
  const int ntTiles = D_ / 64;              // 16
  int tm = (wave / ntTiles) * 16;
  int tn = (wave % ntTiles) * 64;
  int col = lane & 15, half = lane >> 4;

  v8f c[4] = {};
  gemm16x64(X + (size_t)(tm + col) * D_, W + (size_t)(tn + col) * D_, D_, half, c);

#pragma unroll
  for (int nt = 0; nt < 4; ++nt) {
    int e = tn + nt * 16 + col;
    float bv = bias[e];
#pragma unroll
    for (int r = 0; r < 8; ++r)
      out[(size_t)(tm + half * 8 + r) * D_ + e] = c[nt][r] + bv;
  }
}

// ---------------------------------------------------------------------------
extern "C" void kernel_launch(void* const* d_in, const int* in_sizes, int n_in,
                              void* d_out, int out_size, void* d_ws, size_t ws_size,
                              hipStream_t stream) {
  (void)in_sizes; (void)n_in; (void)out_size; (void)ws_size;
  const float* x    = (const float*)d_in[0];
  // d_in[1] is the boolean causal mask; causality is applied analytically.
  const float* Wqkv = (const float*)d_in[2];
  const float* bqkv = (const float*)d_in[3];
  const float* Wo   = (const float*)d_in[4];
  const float* bo   = (const float*)d_in[5];
  float* out = (float*)d_out;

  const size_t headElems = (size_t)B_ * H_ * L_ * HD_;       // 8,388,608
  uint16_t* qkvb    = (uint16_t*)d_ws;                       // bf16 Q|K|V: 48 MB
  float*    attnOut = (float*)((char*)d_ws + 3 * headElems * sizeof(uint16_t)); // 32 MB

  {  // QKV projection: 512 x 48 (16x64) tiles, 8 waves/block
    int waves = (M_ / 16) * (N3_ / 64);                      // 24576
    qkv_gemm_kernel<<<waves / 8, 256, 0, stream>>>(x, Wqkv, bqkv, qkvb);
  }
  {  // Attention: 64 (b,h) x 128 q-tiles = 8192 waves, 4 waves/block
    int waves = B_ * H_ * (L_ / 16);
    attn_kernel<<<waves / 4, 128, 0, stream>>>(qkvb, qkvb + headElems,
                                               qkvb + 2 * headElems, attnOut);
  }
  {  // Output projection: 512 x 16 (16x64) tiles
    int waves = (M_ / 16) * (D_ / 16) / 4;                   // 8192
    out_gemm_kernel<<<waves / 8, 256, 0, stream>>>(attnOut, Wo, bo, out);
  }
}
